// Model_21234318312051
// MI455X (gfx1250) — matrix operand order repaired and checked by reading, not compiled, not run on hardware
//
#include <hip/hip_runtime.h>
#include <hip/hip_bf16.h>

typedef __attribute__((ext_vector_type(16))) _Float16 v16h;
typedef __attribute__((ext_vector_type(8)))  _Float16 v8h;
typedef __attribute__((ext_vector_type(8)))  float    v8f;

// ---------------------------------------------------------------------------
// Model dims
// ---------------------------------------------------------------------------
#define BB 64
#define NN 512

// ---------------------------------------------------------------------------
// Weight pre-swizzle into WMMA B-fragment layout:
// buffer index = (((ntile*nK + kstep)*32 + lane)*16 + i), where for a 32x16
// B tile: lane = (half<<4)|n16, element i holds K = (i>>3)*16 + half*8 + (i&7)  (ISA 7.12.2)
// ---------------------------------------------------------------------------
__global__ void prep_w_kernel(const float* __restrict__ src, _Float16* __restrict__ dst,
                              int K, int N, int Kpad, int Npad) {
  int t = blockIdx.x * blockDim.x + threadIdx.x;
  if (t >= Kpad * Npad) return;
  int kp = t / Npad, np_ = t % Npad;
  int ntile = np_ >> 4, n16 = np_ & 15;
  int kstep = kp >> 5,  k32 = kp & 31;
  int hsel = (k32 >> 3) & 1,  i   = (k32 & 7) | ((k32 >> 4) << 3);
  int lane = (hsel << 4) | n16;
  int nK = Kpad >> 5;
  float v = (kp < K && np_ < N) ? src[(size_t)kp * N + np_] : 0.0f;
  dst[(((size_t)ntile * nK + kstep) * 32 + lane) * 16 + i] = (_Float16)v;
}

// ---------------------------------------------------------------------------
// Generic WMMA GEMM: C[M,N] = act(A[M,K](f32,lda) * Wfrag + bias)
// One block = 64 rows x all N cols, 8 waves, each wave owns 16x16 tiles.
// ---------------------------------------------------------------------------
__global__ void gemm_wmma_kernel(const float* __restrict__ A, int lda,
                                 int K, int N,
                                 const _Float16* __restrict__ Wf,
                                 const float* __restrict__ bias,
                                 float* __restrict__ C, int ldc, int coloff, int act) {
  const int tid  = threadIdx.x;
  const int wave = tid >> 5, lane = tid & 31;
  const int hsel = lane >> 4, l15 = lane & 15;
  const int rowBase = blockIdx.x * 64;
  const int mt = 4, nt = N >> 4, nK = (K + 31) >> 5;
  for (int t = wave; t < mt * nt; t += 8) {
    int mtile = t % mt, ntile = t / mt;
    v8f acc = {};
    int row = rowBase + mtile * 16 + l15;
    const float* Arow = A + (size_t)row * lda;
    for (int kk = 0; kk < nK; ++kk) {
      const int kb = kk * 32;
      v16h a;
      if (kb + 32 <= K) {
        // Full fragment: two runs of 8 contiguous floats, no guards.
        const float* g0 = Arow + kb + hsel * 8;
        const float* g1 = g0 + 16;
#pragma unroll
        for (int i = 0; i < 8; ++i) a[i] = (_Float16)g0[i];
#pragma unroll
        for (int i = 0; i < 8; ++i) a[8 + i] = (_Float16)g1[i];
      } else {
        // Partial tail (only the tiny K=4 layer): clamp address, select value.
#pragma unroll
        for (int i = 0; i < 16; ++i) {
          int kcol = kb + ((i >> 3) << 4) + hsel * 8 + (i & 7);
          float av = Arow[kcol < K ? kcol : 0];
          a[i] = (_Float16)(kcol < K ? av : 0.0f);
        }
      }
      v16h bf = *(const v16h*)(Wf + ((size_t)(ntile * nK + kk) * 32 + lane) * 16);
      acc = __builtin_amdgcn_wmma_f32_16x16x32_f16(false, a, false, bf,
                                                   (short)0, acc, false, false);
    }
    int ncol = ntile * 16 + l15;
    float bv = bias[ncol];
#pragma unroll
    for (int r = 0; r < 8; ++r) {
      float v = acc[r] + bv;
      if (act == 1) v = fmaxf(v, 0.0f);
      else if (act == 2) v = 1.0f / (1.0f + __expf(-v));
      int m = rowBase + mtile * 16 + r + hsel * 8;
      C[(size_t)m * ldc + coloff + ncol] = v;
    }
  }
}

// ---------------------------------------------------------------------------
// LDS->LDS WMMA GEMM + bias + ReLU for the fused SA kernel.
// Input tile stride == Cin (multiple of 32). Output stride == Cout.
// ---------------------------------------------------------------------------
__device__ __forceinline__ void lds_gemm_relu(const _Float16* smIn, int Cin,
                                              const _Float16* __restrict__ Wf,
                                              const float* __restrict__ bias,
                                              _Float16* smOut, int Cout,
                                              int Mr, int tid) {
  const int wave = tid >> 5, lane = tid & 31;
  const int hsel = lane >> 4, l15 = lane & 15;
  const int mt = Mr >> 4, nt = Cout >> 4, nK = Cin >> 5;
  for (int t = wave; t < mt * nt; t += 8) {
    int mtile = t % mt, ntile = t / mt;
    v8f acc = {};
    const _Float16* arow = smIn + (mtile * 16 + l15) * Cin + hsel * 8;
    for (int kk = 0; kk < nK; ++kk) {
      v8h lo = *(const v8h*)(arow + kk * 32);        // K = kb + hsel*8 + 0..7
      v8h hi = *(const v8h*)(arow + kk * 32 + 16);   // K = kb + 16 + hsel*8 + 0..7
      v16h a = __builtin_shufflevector(lo, hi, 0, 1, 2, 3, 4, 5, 6, 7,
                                       8, 9, 10, 11, 12, 13, 14, 15);
      v16h bf = *(const v16h*)(Wf + ((size_t)(ntile * nK + kk) * 32 + lane) * 16);
      acc = __builtin_amdgcn_wmma_f32_16x16x32_f16(false, a, false, bf,
                                                   (short)0, acc, false, false);
    }
    int ncol = ntile * 16 + l15;
    float bv = bias[ncol];
#pragma unroll
    for (int r = 0; r < 8; ++r) {
      float v = fmaxf(acc[r] + bv, 0.0f);
      smOut[(mtile * 16 + r + hsel * 8) * Cout + ncol] = (_Float16)v;
    }
  }
}

// ---------------------------------------------------------------------------
// Fused SA layer: gather+center -> 3x (WMMA GEMM + ReLU) -> max over samples.
// One block per (batch, center). 256 threads = 8 wave32.
// LDS: region0 = 10240 halfs (grouped tile, later layer-2 out),
//      region1 = 16384 halfs (layer-1 out, layer-3 out). Total 53.2 KB.
// ---------------------------------------------------------------------------
__global__ void sa_fused_kernel(const float* __restrict__ xyz, int xstride, int N,
                                const float* __restrict__ feat, int C,
                                const float* __restrict__ new_xyz,
                                const int* __restrict__ idx,
                                int S, int Krows, int C0pad,
                                const _Float16* __restrict__ w0,
                                const _Float16* __restrict__ w1,
                                const _Float16* __restrict__ w2,
                                const float* __restrict__ b0,
                                const float* __restrict__ b1,
                                const float* __restrict__ b2,
                                int C1, int C2, int C3,
                                float* __restrict__ out) {
  __shared__ __align__(16) _Float16 smem[26624];
  _Float16* sm0 = smem;          // 10240 halfs
  _Float16* sm1 = smem + 10240;  // 16384 halfs
  const int blk = blockIdx.x;
  const int b = blk / S;
  const int tid = threadIdx.x;
  const float* ctr = new_xyz + (size_t)blk * 3;
  const int* id = idx + (size_t)blk * Krows;
  // Stage grouped tile [Krows, C0pad] = [xyz - center | feat | 0-pad] as f16.
  // Phases per row: 3 xyz elems, contiguous feat copy, zero tail (no per-elem
  // predicated loads).
  const int lane64 = tid & 63;
  for (int r = tid >> 6; r < Krows; r += 4) {
    int j = id[r];
    const float* px = xyz + ((size_t)b * N + j) * xstride;
    const float* pf = feat + ((size_t)b * N + j) * C;
    _Float16* dst = sm0 + r * C0pad;
    if (lane64 < 3) dst[lane64] = (_Float16)(px[lane64] - ctr[lane64]);
    for (int c = lane64; c < C; c += 64) dst[3 + c] = (_Float16)pf[c];
    for (int c = 3 + C + lane64; c < C0pad; c += 64) dst[c] = (_Float16)0.0f;
  }
  __syncthreads();
  lds_gemm_relu(sm0, C0pad, w0, b0, sm1, C1, Krows, tid);
  __syncthreads();
  lds_gemm_relu(sm1, C1, w1, b1, sm0, C2, Krows, tid);
  __syncthreads();
  lds_gemm_relu(sm0, C2, w2, b2, sm1, C3, Krows, tid);
  __syncthreads();
  for (int c = tid; c < C3; c += 256) {
    float m = -1e30f;
    for (int r = 0; r < Krows; ++r) m = fmaxf(m, (float)sm1[r * C3 + c]);
    out[(size_t)blk * C3 + c] = m;
  }
}

// ---------------------------------------------------------------------------
// Farthest point sampling, one block per batch (matches jax scan semantics:
// first index 0; argmax = first occurrence).
// ---------------------------------------------------------------------------
__global__ void fps_kernel(const float* __restrict__ xyz, int xstride, int N,
                           int npoint, int* __restrict__ idx_out,
                           float* __restrict__ new_xyz) {
  __shared__ float dists[512];
  __shared__ float rv[256];
  __shared__ int ri[256];
  const int b = blockIdx.x, tid = threadIdx.x;
  const float* base = xyz + (size_t)b * N * xstride;
  for (int i = tid; i < N; i += 256) dists[i] = 1e10f;
  if (tid == 0) {
    idx_out[b * npoint] = 0;
    new_xyz[(size_t)b * npoint * 3 + 0] = base[0];
    new_xyz[(size_t)b * npoint * 3 + 1] = base[1];
    new_xyz[(size_t)b * npoint * 3 + 2] = base[2];
  }
  int last = 0;
  __syncthreads();
  for (int s = 1; s < npoint; ++s) {
    float px = base[(size_t)last * xstride + 0];
    float py = base[(size_t)last * xstride + 1];
    float pz = base[(size_t)last * xstride + 2];
    float bestd = -1.0f; int besti = 0;
    for (int i = tid; i < N; i += 256) {
      const float* q = base + (size_t)i * xstride;
      float dx = q[0] - px, dy = q[1] - py, dz = q[2] - pz;
      float nd = fminf(dists[i], dx * dx + dy * dy + dz * dz);
      dists[i] = nd;
      if (nd > bestd) { bestd = nd; besti = i; }
    }
    rv[tid] = bestd; ri[tid] = besti;
    __syncthreads();
    for (int off = 128; off > 0; off >>= 1) {
      if (tid < off) {
        if (rv[tid + off] > rv[tid] ||
            (rv[tid + off] == rv[tid] && ri[tid + off] < ri[tid])) {
          rv[tid] = rv[tid + off]; ri[tid] = ri[tid + off];
        }
      }
      __syncthreads();
    }
    last = ri[0];
    if (tid == 0) {
      idx_out[b * npoint + s] = last;
      new_xyz[((size_t)b * npoint + s) * 3 + 0] = base[(size_t)last * xstride + 0];
      new_xyz[((size_t)b * npoint + s) * 3 + 1] = base[(size_t)last * xstride + 1];
      new_xyz[((size_t)b * npoint + s) * 3 + 2] = base[(size_t)last * xstride + 2];
    }
    __syncthreads();
  }
}

// ---------------------------------------------------------------------------
// Ball query (CUDA semantics: first <=K in-radius indices; pad with first; 0 if none)
// One thread per center.
// ---------------------------------------------------------------------------
__global__ void ball_kernel(const float* __restrict__ xyz, int xstride, int N,
                            const float* __restrict__ centers, int S, int total,
                            float r2, int K, int* __restrict__ out) {
  int t = blockIdx.x * blockDim.x + threadIdx.x;
  if (t >= total) return;
  int b = t / S;
  const float* base = xyz + (size_t)b * N * xstride;
  float cx = centers[(size_t)t * 3 + 0];
  float cy = centers[(size_t)t * 3 + 1];
  float cz = centers[(size_t)t * 3 + 2];
  int cnt = 0, first = 0;
  int* o = out + (size_t)t * K;
  for (int j = 0; j < N; ++j) {
    const float* q = base + (size_t)j * xstride;
    float dx = q[0] - cx, dy = q[1] - cy, dz = q[2] - cz;
    if (dx * dx + dy * dy + dz * dz < r2) {
      if (cnt == 0) first = j;
      if (cnt < K) o[cnt] = j;
      cnt++;
    }
  }
  if (cnt == 0) {
    for (int k = 0; k < K; ++k) o[k] = 0;
  } else {
    for (int k = (cnt < K ? cnt : K); k < K; ++k) o[k] = first;
  }
}

// featA[:, 0:64] = x[:, 7:71]
__global__ void copy_extra_kernel(const float* __restrict__ x, float* __restrict__ featA) {
  int t = blockIdx.x * blockDim.x + threadIdx.x;
  if (t >= BB * NN * 64) return;
  int c = t & 63; int p = t >> 6;
  featA[(size_t)p * 128 + c] = x[(size_t)p * 71 + 7 + c];
}

// SA3 "grouping": identity indices over the 32 remaining points, zero centers.
__global__ void init_sa3_kernel(int* __restrict__ ball3, float* __restrict__ nx3) {
  int t = blockIdx.x * blockDim.x + threadIdx.x;
  if (t < BB * 32) ball3[t] = t & 31;
  if (t < BB * 3) nx3[t] = 0.0f;
}

// Tiny final FC layers (N=1 or 7): plain GEMV, act 0=none, 2=sigmoid.
__global__ void gemv_kernel(const float* __restrict__ A, int lda,
                            const float* __restrict__ W, const float* __restrict__ bias,
                            int M, int K, int N, int act, float* __restrict__ out) {
  int t = blockIdx.x * blockDim.x + threadIdx.x;
  if (t >= M * N) return;
  int m = t / N, n = t % N;
  float s = bias[n];
  const float* a = A + (size_t)m * lda;
  for (int k = 0; k < K; ++k) s += a[k] * W[(size_t)k * N + n];
  if (act == 2) s = 1.0f / (1.0f + __expf(-s));
  out[(size_t)m * N + n] = s;
}

// ---------------------------------------------------------------------------
extern "C" void kernel_launch(void* const* d_in, const int* in_sizes, int n_in,
                              void* d_out, int out_size, void* d_ws, size_t ws_size,
                              hipStream_t stream) {
  (void)in_sizes; (void)n_in; (void)out_size; (void)ws_size;
  const float* x = (const float*)d_in[0];
  const float* local_w[2] = { (const float*)d_in[1], (const float*)d_in[2] };
  const float* local_b[2] = { (const float*)d_in[3], (const float*)d_in[4] };
  const float* sa_w[9]; for (int i = 0; i < 9; ++i) sa_w[i] = (const float*)d_in[5 + i];
  const float* sa_b[9]; for (int i = 0; i < 9; ++i) sa_b[i] = (const float*)d_in[14 + i];
  const float* cls_w[3] = { (const float*)d_in[23], (const float*)d_in[24], (const float*)d_in[25] };
  const float* cls_b[3] = { (const float*)d_in[26], (const float*)d_in[27], (const float*)d_in[28] };
  const float* det_w[3] = { (const float*)d_in[29], (const float*)d_in[30], (const float*)d_in[31] };
  const float* det_b[3] = { (const float*)d_in[32], (const float*)d_in[33], (const float*)d_in[34] };
  float* outp = (float*)d_out;

  // ---- workspace layout ----
  char* ws = (char*)d_ws;
  size_t off = 0;
  auto alloc = [&](size_t bytes) -> char* {
    char* p = ws + off;
    off += (bytes + 255) & ~(size_t)255;
    return p;
  };
  float* featA  = (float*)alloc((size_t)BB * NN * 128 * 4);
  float* hloc   = (float*)alloc((size_t)BB * NN * 64 * 4);
  float* feat1  = (float*)alloc((size_t)BB * 128 * 128 * 4);
  float* feat2  = (float*)alloc((size_t)BB * 32 * 256 * 4);
  float* feat3  = (float*)alloc((size_t)BB * 512 * 4);
  float* nx1    = (float*)alloc((size_t)BB * 128 * 3 * 4);
  float* nx2    = (float*)alloc((size_t)BB * 32 * 3 * 4);
  float* nx3    = (float*)alloc((size_t)BB * 3 * 4);
  int*   idx1   = (int*)alloc((size_t)BB * 128 * 4);
  int*   idx2   = (int*)alloc((size_t)BB * 32 * 4);
  int*   ball1  = (int*)alloc((size_t)BB * 128 * 64 * 4);
  int*   ball2  = (int*)alloc((size_t)BB * 32 * 64 * 4);
  int*   ball3  = (int*)alloc((size_t)BB * 32 * 4);
  float* hc1    = (float*)alloc((size_t)BB * 256 * 4);
  float* hc2    = (float*)alloc((size_t)BB * 256 * 4);
  float* hd1    = (float*)alloc((size_t)BB * 256 * 4);
  float* hd2    = (float*)alloc((size_t)BB * 256 * 4);

  // Pre-swizzled f16 weight fragments
  struct WSpec { const float* src; int K, N, Kpad, Npad; _Float16* dst; };
  WSpec wspec[15] = {
    { local_w[0],   4,  64,  32,  64, nullptr },
    { local_w[1],  64,  64,  64,  64, nullptr },
    { sa_w[0],    131, 128, 160, 128, nullptr },
    { sa_w[1],    128, 128, 128, 128, nullptr },
    { sa_w[2],    128, 128, 128, 128, nullptr },
    { sa_w[3],    131, 128, 160, 128, nullptr },
    { sa_w[4],    128, 128, 128, 128, nullptr },
    { sa_w[5],    128, 256, 128, 256, nullptr },
    { sa_w[6],    259, 256, 288, 256, nullptr },
    { sa_w[7],    256, 256, 256, 256, nullptr },
    { sa_w[8],    256, 512, 256, 512, nullptr },
    { cls_w[0],   512, 256, 512, 256, nullptr },
    { cls_w[1],   256, 256, 256, 256, nullptr },
    { det_w[0],   512, 256, 512, 256, nullptr },
    { det_w[1],   256, 256, 256, 256, nullptr },
  };
  for (int i = 0; i < 15; ++i)
    wspec[i].dst = (_Float16*)alloc((size_t)wspec[i].Kpad * wspec[i].Npad * 2);

  for (int i = 0; i < 15; ++i) {
    int total = wspec[i].Kpad * wspec[i].Npad;
    prep_w_kernel<<<(total + 255) / 256, 256, 0, stream>>>(
        wspec[i].src, wspec[i].dst, wspec[i].K, wspec[i].N, wspec[i].Kpad, wspec[i].Npad);
  }

  _Float16* fL0 = wspec[0].dst, *fL1 = wspec[1].dst;
  _Float16* fS1[3] = { wspec[2].dst, wspec[3].dst, wspec[4].dst };
  _Float16* fS2[3] = { wspec[5].dst, wspec[6].dst, wspec[7].dst };
  _Float16* fS3[3] = { wspec[8].dst, wspec[9].dst, wspec[10].dst };
  _Float16* fC0 = wspec[11].dst, *fC1 = wspec[12].dst;
  _Float16* fD0 = wspec[13].dst, *fD1 = wspec[14].dst;

  // ---- 1) local MLP + feature assembly: feat = [x[...,7:71] | relu MLP(x[...,3:7])] ----
  copy_extra_kernel<<<(BB * NN * 64 + 255) / 256, 256, 0, stream>>>(x, featA);
  gemm_wmma_kernel<<<(BB * NN) / 64, 256, 0, stream>>>(x + 3, 71, 4, 64, fL0, local_b[0],
                                                       hloc, 64, 0, 1);
  gemm_wmma_kernel<<<(BB * NN) / 64, 256, 0, stream>>>(hloc, 64, 64, 64, fL1, local_b[1],
                                                       featA, 128, 64, 1);

  // ---- 2) SA layer 1 (npoint=128, r=0.5, nsample=64, 131->128->128->128) ----
  fps_kernel<<<BB, 256, 0, stream>>>(x, 71, NN, 128, idx1, nx1);
  ball_kernel<<<(BB * 128 + 255) / 256, 256, 0, stream>>>(x, 71, NN, nx1, 128, BB * 128,
                                                          0.25f, 64, ball1);
  sa_fused_kernel<<<BB * 128, 256, 0, stream>>>(x, 71, NN, featA, 128, nx1, ball1,
                                                128, 64, 160,
                                                fS1[0], fS1[1], fS1[2],
                                                sa_b[0], sa_b[1], sa_b[2],
                                                128, 128, 128, feat1);

  // ---- 3) SA layer 2 (npoint=32, r=1.0, nsample=64, 131->128->128->256) ----
  fps_kernel<<<BB, 256, 0, stream>>>(nx1, 3, 128, 32, idx2, nx2);
  ball_kernel<<<(BB * 32 + 255) / 256, 256, 0, stream>>>(nx1, 3, 128, nx2, 32, BB * 32,
                                                         1.0f, 64, ball2);
  sa_fused_kernel<<<BB * 32, 256, 0, stream>>>(nx1, 3, 128, feat1, 128, nx2, ball2,
                                               32, 64, 160,
                                               fS2[0], fS2[1], fS2[2],
                                               sa_b[3], sa_b[4], sa_b[5],
                                               128, 128, 256, feat2);

  // ---- 4) SA layer 3 (global: all 32 points, raw xyz, 259->256->256->512) ----
  init_sa3_kernel<<<(BB * 32 + 255) / 256, 256, 0, stream>>>(ball3, nx3);
  sa_fused_kernel<<<BB, 256, 0, stream>>>(nx2, 3, 32, feat2, 256, nx3, ball3,
                                          1, 32, 288,
                                          fS3[0], fS3[1], fS3[2],
                                          sa_b[6], sa_b[7], sa_b[8],
                                          256, 256, 512, feat3);

  // ---- 5) heads (M=64) ----
  gemm_wmma_kernel<<<1, 256, 0, stream>>>(feat3, 512, 512, 256, fC0, cls_b[0], hc1, 256, 0, 1);
  gemm_wmma_kernel<<<1, 256, 0, stream>>>(hc1, 256, 256, 256, fC1, cls_b[1], hc2, 256, 0, 1);
  gemm_wmma_kernel<<<1, 256, 0, stream>>>(feat3, 512, 512, 256, fD0, det_b[0], hd1, 256, 0, 1);
  gemm_wmma_kernel<<<1, 256, 0, stream>>>(hd1, 256, 256, 256, fD1, det_b[1], hd2, 256, 0, 1);

  // pred_box [64,7] then pred_class [64,1], concatenated flat
  gemv_kernel<<<(BB * 7 + 255) / 256, 256, 0, stream>>>(hd2, 256, det_w[2], det_b[2],
                                                        BB, 256, 7, 0, outp);
  gemv_kernel<<<(BB * 1 + 255) / 256, 256, 0, stream>>>(hc2, 256, cls_w[2], cls_b[2],
                                                        BB, 256, 1, 2, outp + BB * 7);
}
